// TFDecepticon_12154757447887
// MI455X (gfx1250) — matrix operand order repaired and checked by reading, not compiled, not run on hardware
//
#include <hip/hip_runtime.h>
#include <hip/hip_bf16.h>

typedef __attribute__((ext_vector_type(16))) __bf16 bf16x16;
typedef __attribute__((ext_vector_type(8)))  __bf16 bf16x8;
typedef __attribute__((ext_vector_type(8)))  float  f32x8;

#define EPSN 1e-5f

static __device__ inline __bf16 f2bf(float f) {
    __hip_bfloat16 h = __float2bfloat16(f);
    return *reinterpret_cast<__bf16*>(&h);
}

static __device__ inline bf16x16 cat8(bf16x8 lo, bf16x8 hi) {
    bf16x16 r;
#pragma unroll
    for (int i = 0; i < 8; ++i) { r[i] = lo[i]; r[i + 8] = hi[i]; }
    return r;
}

// A-fragment: 16x32 bf16 (M x K). Row-major source, row stride `stride` elems.
// ISA layout: lanes 0-15 row M=lane, K {kk+0..7} in v0..3 and {kk+16..23} in v4..7;
//             lanes 16-31 row M=lane-16, K {kk+8..15} and {kk+24..31}.
static __device__ inline bf16x16 load_frag_a(const __bf16* src, int stride,
                                             int rowBase, int kk, int lane) {
    const int hl = lane >> 4;
    const __bf16* r = src + (size_t)(rowBase + (lane & 15)) * stride;
    bf16x8 lo = *(const bf16x8*)(r + kk + hl * 8);
    bf16x8 hi = *(const bf16x8*)(r + kk + 16 + hl * 8);
    return cat8(lo, hi);
}

// B-fragment: 32x16 bf16 (K x N); column n of B is row (rowBase+n) of src.
// ISA layout: lanes 0-15 col N=lane hold K {kk+0..7} and {kk+16..23}; lanes 16-31 hold K {kk+8..15} and {kk+24..31}.
static __device__ inline bf16x16 load_frag_b(const __bf16* src, int stride,
                                             int rowBase, int kk, int lane) {
    const int hl = lane >> 4;
    const __bf16* r = src + (size_t)(rowBase + (lane & 15)) * stride;
    bf16x8 lo = *(const bf16x8*)(r + kk + hl * 8);
    bf16x8 hi = *(const bf16x8*)(r + kk + 16 + hl * 8);
    return cat8(lo, hi);
}

static __device__ inline f32x8 wmma_bf16(bf16x16 a, bf16x16 b, f32x8 c) {
    return __builtin_amdgcn_wmma_f32_16x16x32_bf16(false, a, false, b,
                                                   (short)0, c, false, false);
}

// Async 16-byte global -> LDS copy (CDNA5 ASYNCcnt path).
// LDS byte offset is the low 32 bits of the generic pointer.
static __device__ inline void async_ld16(const __bf16* g, __bf16* l) {
    unsigned laddr = (unsigned)(uintptr_t)l;
    asm volatile("global_load_async_to_lds_b128 %0, %1, off"
                 :: "v"(laddr), "v"((unsigned long long)(uintptr_t)g)
                 : "memory");
}

static __device__ inline void wait_async0() {
    asm volatile("s_wait_asynccnt 0x0" ::: "memory");
}

// ---------------------------------------------------------------------------
// K0: convert lw (64x64 f32) -> bf16
// ---------------------------------------------------------------------------
__global__ void cvt_lw_kernel(const float* __restrict__ lw, __bf16* __restrict__ lwb) {
    int i = blockIdx.x * 256 + threadIdx.x;
    if (i < 4096) lwb[i] = f2bf(lw[i]);
}

// ---------------------------------------------------------------------------
// K1: fused QKV projection + PReLU + per-(head,b,t) layernorm + affine.
// One block per (b,t). Writes bf16 Q/K row-major [n][t][256], V transposed [n][d][t].
// ---------------------------------------------------------------------------
__global__ __launch_bounds__(256)
void qkv_kernel(const float* __restrict__ x,
                const float* __restrict__ qw, const float* __restrict__ qb,
                const float* __restrict__ qa, const float* __restrict__ qg,
                const float* __restrict__ qbe,
                const float* __restrict__ kw, const float* __restrict__ kb,
                const float* __restrict__ ka, const float* __restrict__ kg,
                const float* __restrict__ kbe,
                const float* __restrict__ vw, const float* __restrict__ vb,
                const float* __restrict__ va, const float* __restrict__ vg,
                const float* __restrict__ vbe,
                __bf16* __restrict__ Qb, __bf16* __restrict__ Kb,
                __bf16* __restrict__ Vt) {
    const int bt = blockIdx.x;
    const int b = bt >> 10, t = bt & 1023;
    const int tid = threadIdx.x;

    __shared__ float xs[64][64];   // x[b,:,t,:]
    __shared__ float ys[96][64];   // rows 0-15 Q, 16-31 K, 32-95 V
    __shared__ float stMu[12], stR[12];

#pragma unroll
    for (int k = 0; k < 16; ++k) {
        int idx = tid + k * 256;
        int c = idx >> 6, f = idx & 63;
        xs[c][f] = x[(((size_t)b * 64 + c) * 1024 + t) * 64 + f];
    }
    __syncthreads();

    const int f = tid & 63;
    const int rr = tid >> 6;
    for (int j = 0; j < 24; ++j) {
        int r = rr + 4 * j;
        const float* wrow; float bias, slope;
        if (r < 16)      { wrow = qw + r * 64;        bias = qb[r];      slope = qa[r >> 2]; }
        else if (r < 32) { wrow = kw + (r - 16) * 64; bias = kb[r - 16]; slope = ka[(r - 16) >> 2]; }
        else             { wrow = vw + (r - 32) * 64; bias = vb[r - 32]; slope = va[(r - 32) >> 4]; }
        float acc = bias;
#pragma unroll 8
        for (int c = 0; c < 64; ++c) acc = fmaf(wrow[c], xs[c][f], acc);
        ys[r][f] = (acc > 0.0f) ? acc : slope * acc;  // PReLU
    }
    __syncthreads();

    const int lane = tid & 31, wv = tid >> 5;
    {   // Q (waves 0-3) / K (waves 4-7) stats: 256 values per group
        int hh = wv & 3;
        int rbase = (wv < 4) ? hh * 4 : 16 + hh * 4;
        float s = 0.f, ss = 0.f;
#pragma unroll
        for (int v = 0; v < 8; ++v) {
            int idx = lane * 8 + v;
            float y = ys[rbase + (idx >> 6)][idx & 63];
            s += y; ss += y * y;
        }
#pragma unroll
        for (int m = 16; m > 0; m >>= 1) {
            s += __shfl_xor(s, m, 32);
            ss += __shfl_xor(ss, m, 32);
        }
        if (lane == 0) {
            float mu = s * (1.0f / 256.0f);
            float var = ss * (1.0f / 256.0f) - mu * mu;
            int g = (wv < 4) ? hh : 4 + hh;
            stMu[g] = mu; stR[g] = rsqrtf(var + EPSN);
        }
    }
    if (wv < 4) {   // V stats: 1024 values per head
        float s = 0.f, ss = 0.f;
#pragma unroll
        for (int v = 0; v < 32; ++v) {
            int idx = lane * 32 + v;
            float y = ys[32 + wv * 16 + (idx >> 6)][idx & 63];
            s += y; ss += y * y;
        }
#pragma unroll
        for (int m = 16; m > 0; m >>= 1) {
            s += __shfl_xor(s, m, 32);
            ss += __shfl_xor(ss, m, 32);
        }
        if (lane == 0) {
            float mu = s * (1.0f / 1024.0f);
            float var = ss * (1.0f / 1024.0f) - mu * mu;
            stMu[8 + wv] = mu; stR[8 + wv] = rsqrtf(var + EPSN);
        }
    }
    __syncthreads();

    for (int j = 0; j < 24; ++j) {
        int r = rr + 4 * j;
        float y = ys[r][f];
        if (r < 16) {
            int hh = r >> 2, hid = r & 3;
            float v = (y - stMu[hh]) * stR[hh] * qg[r * 64 + f] + qbe[r * 64 + f];
            Qb[(((size_t)(hh * 8 + b)) * 1024 + t) * 256 + hid * 64 + f] = f2bf(v);
        } else if (r < 32) {
            int r2 = r - 16, hh = r2 >> 2, hid = r2 & 3;
            float v = (y - stMu[4 + hh]) * stR[4 + hh] * kg[r2 * 64 + f] + kbe[r2 * 64 + f];
            Kb[(((size_t)(hh * 8 + b)) * 1024 + t) * 256 + hid * 64 + f] = f2bf(v);
        } else {
            int r3 = r - 32, hh = r3 >> 4, dv = r3 & 15;
            float v = (y - stMu[8 + hh]) * stR[8 + hh] * vg[r3 * 64 + f] + vbe[r3 * 64 + f];
            // transposed: Vt[n][d = dv*64+f][t]
            Vt[(((size_t)(hh * 8 + b)) * 1024 + (dv * 64 + f)) * 1024 + t] = f2bf(v);
        }
    }
}

// ---------------------------------------------------------------------------
// K2: flash attention. Block = (n, 32-row query tile), 8 waves.
// Wave w: row-tile rt = w>>2 (16 rows), col slot ct = w&3 (S key-tile / PV 256-col group).
// K tile (64x256 bf16, 32KB) is staged in LDS with async global->LDS copies,
// software pipelined one iteration ahead (copy overlaps softmax + PV wmma).
// ---------------------------------------------------------------------------
__global__ __launch_bounds__(256)
void attn_kernel(const __bf16* __restrict__ Qb, const __bf16* __restrict__ Kb,
                 const __bf16* __restrict__ Vt, __bf16* __restrict__ Aout) {
    const int tid = threadIdx.x;
    const int lane = tid & 31;
    const int wv = tid >> 5;
    const int rt = wv >> 2;
    const int ct = wv & 3;
    const int n = blockIdx.y;
    const int t0 = blockIdx.x * 32;
    const int h = n >> 3, b = n & 7;
    const int hl = lane >> 4, col = lane & 15;

    const __bf16* Qn = Qb + (size_t)n * 1024 * 256;
    const __bf16* Kn = Kb + (size_t)n * 1024 * 256;
    const __bf16* Vn = Vt + (size_t)n * 1024 * 1024;

    __shared__ float  S[32][64];
    __shared__ __bf16 P[32][64];
    __shared__ __bf16 Ks[64][256];   // staged K tile (keys x D)
    __shared__ float  alpha_s[32];
    __shared__ float  l_s[32];

    // resident Q fragments for this wave's 16-row tile (D = 256 -> 8 K-chunks)
    bf16x16 aQ[8];
#pragma unroll
    for (int kc = 0; kc < 8; ++kc)
        aQ[kc] = load_frag_a(Qn, 256, t0 + rt * 16, kc * 32, lane);

    f32x8 acc[16] = {};
    float m_run = -3.0e38f, l_run = 0.0f;
    const float scale = 0.0625f;  // 1/sqrt(HID*F=256)

    // issue async copy of K tile 0 (each thread: 8 x 16B chunks)
#pragma unroll
    for (int q = 0; q < 8; ++q) {
        int chunk = tid + q * 256;            // 2048 chunks = 64 rows x 32
        int row = chunk >> 5, co = (chunk & 31) * 8;
        async_ld16(Kn + (size_t)row * 256 + co, &Ks[row][co]);
    }

    for (int it = 0; it < 16; ++it) {
        const int s0 = it * 64;
        wait_async0();
        __syncthreads();                      // K tile `it` visible in LDS

        // --- S tile: rows [rt*16,+16) x keys [s0+ct*16,+16), K-dim 256
        f32x8 sAcc = {};
#pragma unroll
        for (int kc = 0; kc < 8; ++kc) {
            bf16x16 bK = load_frag_b(&Ks[0][0], 256, ct * 16, kc * 32, lane);
            sAcc = wmma_bf16(aQ[kc], bK, sAcc);
        }
#pragma unroll
        for (int r = 0; r < 8; ++r)
            S[rt * 16 + r + hl * 8][ct * 16 + col] = sAcc[r] * scale;
        __syncthreads();                      // S ready; all Ks reads done

        // prefetch next K tile; overlaps softmax + PV below
        if (it < 15) {
#pragma unroll
            for (int q = 0; q < 8; ++q) {
                int chunk = tid + q * 256;
                int row = chunk >> 5, co = (chunk & 31) * 8;
                async_ld16(Kn + (size_t)(s0 + 64 + row) * 256 + co, &Ks[row][co]);
            }
        }

        // --- online softmax, one thread per row
        if (tid < 32) {
            int row = tid;
            float mt = -3.0e38f;
#pragma unroll 8
            for (int j = 0; j < 64; ++j) mt = fmaxf(mt, S[row][j]);
            float mn = fmaxf(m_run, mt);
            float al = __expf(m_run - mn);
            float ls = 0.0f;
#pragma unroll 8
            for (int j = 0; j < 64; ++j) {
                float p = __expf(S[row][j] - mn);
                P[row][j] = f2bf(p);
                ls += p;
            }
            l_run = l_run * al + ls;
            m_run = mn;
            alpha_s[row] = al;
            if (it == 15) l_s[row] = l_run;
        }
        __syncthreads();

        // --- rescale accumulators by alpha(row)
        float al[8];
#pragma unroll
        for (int r = 0; r < 8; ++r) al[r] = alpha_s[rt * 16 + r + hl * 8];
#pragma unroll
        for (int i = 0; i < 16; ++i)
#pragma unroll
            for (int r = 0; r < 8; ++r) acc[i][r] *= al[r];

        // --- P x V: this wave's 256-col group (16 tiles), 64-key K-dim
        bf16x16 p0 = load_frag_a(&P[0][0], 64, rt * 16, 0, lane);
        bf16x16 p1 = load_frag_a(&P[0][0], 64, rt * 16, 32, lane);
#pragma unroll
        for (int i = 0; i < 16; ++i) {
            int cb = ct * 256 + i * 16;
            bf16x16 v0 = load_frag_b(Vn, 1024, cb, s0 + 0, lane);
            bf16x16 v1 = load_frag_b(Vn, 1024, cb, s0 + 32, lane);
            acc[i] = wmma_bf16(p0, v0, acc[i]);
            acc[i] = wmma_bf16(p1, v1, acc[i]);
        }
    }

    // --- epilogue: O = acc / l, write to Aout[b][t][f][c] (c = h*16+dv fastest)
    float li[8];
#pragma unroll
    for (int r = 0; r < 8; ++r) li[r] = 1.0f / l_s[rt * 16 + r + hl * 8];
#pragma unroll
    for (int i = 0; i < 16; ++i) {
        int d = ct * 256 + i * 16 + col;
        int dv = d >> 6, f = d & 63;
        int c = h * 16 + dv;
#pragma unroll
        for (int r = 0; r < 8; ++r) {
            int t = t0 + rt * 16 + r + hl * 8;
            Aout[(((size_t)b * 1024 + t) * 64 + f) * 64 + c] = f2bf(acc[i][r] * li[r]);
        }
    }
}

// ---------------------------------------------------------------------------
// K3: final conv block (64x64 lw @ A) + PReLU + layernorm + affine + residual.
// One block per (b,t); 16 output tiles, 2 per wave, K = 64.
// ---------------------------------------------------------------------------
__global__ __launch_bounds__(256)
void final_kernel(const __bf16* __restrict__ Aout, const __bf16* __restrict__ lwb,
                  const float* __restrict__ lb, const float* __restrict__ lap,
                  const float* __restrict__ lg, const float* __restrict__ lbe,
                  const float* __restrict__ x, float* __restrict__ out) {
    const int bt = blockIdx.x;
    const int b = bt >> 10, t = bt & 1023;
    const int lane = threadIdx.x & 31, wv = threadIdx.x >> 5;
    const int hl = lane >> 4, col = lane & 15;
    const float la = lap[0];

    const __bf16* Ab = Aout + (size_t)bt * 4096;  // [f][c], 64x64

    __shared__ float redS[256], redQ[256];

    const int mt0 = wv >> 2, nt0 = wv & 3;
    const int mt1 = (wv + 8) >> 2, nt1 = (wv + 8) & 3;
    f32x8 acc0 = {}, acc1 = {};
#pragma unroll
    for (int kc = 0; kc < 2; ++kc) {
        int kk = kc * 32;
        bf16x16 a0 = load_frag_a(lwb, 64, mt0 * 16, kk, lane);
        bf16x16 b0 = load_frag_b(Ab, 64, nt0 * 16, kk, lane);
        acc0 = wmma_bf16(a0, b0, acc0);
        bf16x16 a1 = load_frag_a(lwb, 64, mt1 * 16, kk, lane);
        bf16x16 b1 = load_frag_b(Ab, 64, nt1 * 16, kk, lane);
        acc1 = wmma_bf16(a1, b1, acc1);
    }

    float y0[8], y1[8];
    float s = 0.f, ss = 0.f;
#pragma unroll
    for (int r = 0; r < 8; ++r) {
        int co0 = mt0 * 16 + r + hl * 8;
        float y = acc0[r] + lb[co0];
        y = (y > 0.0f) ? y : la * y;
        y0[r] = y; s += y; ss += y * y;
        int co1 = mt1 * 16 + r + hl * 8;
        float z = acc1[r] + lb[co1];
        z = (z > 0.0f) ? z : la * z;
        y1[r] = z; s += z; ss += z * z;
    }
    redS[threadIdx.x] = s;
    redQ[threadIdx.x] = ss;
    __syncthreads();
#pragma unroll
    for (int st = 128; st > 0; st >>= 1) {
        if (threadIdx.x < st) {
            redS[threadIdx.x] += redS[threadIdx.x + st];
            redQ[threadIdx.x] += redQ[threadIdx.x + st];
        }
        __syncthreads();
    }
    float mu = redS[0] * (1.0f / 4096.0f);
    float var = redQ[0] * (1.0f / 4096.0f) - mu * mu;
    float rin = rsqrtf(var + EPSN);

#pragma unroll
    for (int r = 0; r < 8; ++r) {
        {
            int co = mt0 * 16 + r + hl * 8, ff = nt0 * 16 + col;
            size_t oi = (((size_t)b * 64 + co) * 1024 + t) * 64 + ff;
            out[oi] = (y0[r] - mu) * rin * lg[co * 64 + ff] + lbe[co * 64 + ff] + x[oi];
        }
        {
            int co = mt1 * 16 + r + hl * 8, ff = nt1 * 16 + col;
            size_t oi = (((size_t)b * 64 + co) * 1024 + t) * 64 + ff;
            out[oi] = (y1[r] - mu) * rin * lg[co * 64 + ff] + lbe[co * 64 + ff] + x[oi];
        }
    }
}

// ---------------------------------------------------------------------------
extern "C" void kernel_launch(void* const* d_in, const int* in_sizes, int n_in,
                              void* d_out, int out_size, void* d_ws, size_t ws_size,
                              hipStream_t stream) {
    (void)in_sizes; (void)n_in; (void)out_size; (void)ws_size;
    const float* x   = (const float*)d_in[0];
    const float* qw  = (const float*)d_in[1];
    const float* qb  = (const float*)d_in[2];
    const float* qa  = (const float*)d_in[3];
    const float* qg  = (const float*)d_in[4];
    const float* qbe = (const float*)d_in[5];
    const float* kw  = (const float*)d_in[6];
    const float* kb  = (const float*)d_in[7];
    const float* ka  = (const float*)d_in[8];
    const float* kg  = (const float*)d_in[9];
    const float* kbe = (const float*)d_in[10];
    const float* vw  = (const float*)d_in[11];
    const float* vb  = (const float*)d_in[12];
    const float* va  = (const float*)d_in[13];
    const float* vg  = (const float*)d_in[14];
    const float* vbe = (const float*)d_in[15];
    const float* lw  = (const float*)d_in[16];
    const float* lb  = (const float*)d_in[17];
    const float* la  = (const float*)d_in[18];
    const float* lg  = (const float*)d_in[19];
    const float* lbe = (const float*)d_in[20];

    __bf16* Qb_  = (__bf16*)d_ws;
    __bf16* Kb_  = Qb_ + (size_t)32 * 1024 * 256;
    __bf16* Vt_  = Kb_ + (size_t)32 * 1024 * 256;
    __bf16* Ao_  = Vt_ + (size_t)32 * 1024 * 1024;
    __bf16* lwb_ = Ao_ + (size_t)8 * 1024 * 64 * 64;

    cvt_lw_kernel<<<16, 256, 0, stream>>>(lw, lwb_);
    qkv_kernel<<<8192, 256, 0, stream>>>(x, qw, qb, qa, qg, qbe,
                                         kw, kb, ka, kg, kbe,
                                         vw, vb, va, vg, vbe,
                                         Qb_, Kb_, Vt_);
    attn_kernel<<<dim3(32, 32), 256, 0, stream>>>(Qb_, Kb_, Vt_, Ao_);
    final_kernel<<<8192, 256, 0, stream>>>(Ao_, lwb_, lb, la, lg, lbe,
                                           x, (float*)d_out);
}
